// Attention_47605417508944
// MI455X (gfx1250) — matrix operand order repaired and checked by reading, not compiled, not run on hardware
//
#include <hip/hip_runtime.h>
#include <hip/hip_bf16.h>

typedef __attribute__((ext_vector_type(16))) __bf16 v16bf;
typedef __attribute__((ext_vector_type(8)))  __bf16 v8bf;
typedef __attribute__((ext_vector_type(8)))  float  v8f;

#define NB   4
#define CIN  256
#define CQ   64
#define HW   4096
#define ALPHA 100.0f

// ---- workspace layout (bytes) ----
#define OFF_YA     (0ull)                       // 4 MiB fp32 conv/feat buffer A
#define OFF_YB     (4ull  * 1024 * 1024)        // 4 MiB fp32 conv/feat buffer B
#define OFF_FAP    (8ull  * 1024 * 1024)        // 2 MiB bf16 fa packed [n][pos][64]
#define OFF_FBP    (10ull * 1024 * 1024)        // 2 MiB bf16 fb packed [n][pos][64]
#define OFF_FCB    (12ull * 1024 * 1024)        // 8 MiB bf16 fc [n][256][4096]
#define OFF_STATS  (20ull * 1024 * 1024)        // fp32 stats block
// stats block (floats): sumA[256] sumsqA[256] sumzA[256] sumB[256] sumsqB[256] sumzB[256] colsum[16384]
#define STATS_FLOATS (6 * NB * CQ + NB * HW)

static __device__ inline __bf16 f2bf(float f) { return (__bf16)f; }

// ================= Tensor Data Mover support =================
#if defined(__has_builtin)
#if __has_builtin(__builtin_amdgcn_tensor_load_to_lds)
#define HAVE_TDM 1
#endif
#endif

#ifdef HAVE_TDM
typedef unsigned int u32x4 __attribute__((ext_vector_type(4)));
typedef int          i32x4 __attribute__((ext_vector_type(4)));
typedef int          i32x8 __attribute__((ext_vector_type(8)));

// DMA a 32(k) x 64(q) fp32 tile of corr (row stride HW floats) into LDS at ldsOff.
// D# built per CDNA5 ISA 8.3/8.4: group0 = {count=1 | lds_addr | global_addr | type=2},
// group1 = {data_size=4B, tensor_dim0/1=HW, tile_dim0=64, tile_dim1=32, dim0_stride=HW}.
static __device__ inline void tdm_load_corr_tile(const float* gsrc, unsigned ldsOff) {
    unsigned long long ga = (unsigned long long)(size_t)gsrc;
    u32x4 g0;
    g0[0] = 1u;                                            // count=1, user mode
    g0[1] = ldsOff;                                        // lds_addr (bytes)
    g0[2] = (unsigned)(ga & 0xFFFFFFFFu);                  // global_addr[31:0]
    g0[3] = (unsigned)((ga >> 32) & 0x01FFFFFFu) | (2u << 30);  // addr[56:32] | type=2
    i32x8 g1;
    g1[0] = (int)(2u << 16);                               // data_size = 4 bytes
    g1[1] = (int)(((unsigned)HW & 0xFFFFu) << 16);         // tensor_dim0[15:0]
    g1[2] = (int)(((unsigned)HW >> 16) | (((unsigned)HW & 0xFFFFu) << 16)); // dim0 hi | dim1 lo
    g1[3] = (int)(((unsigned)HW >> 16) | (64u << 16));     // dim1 hi | tile_dim0 = 64
    g1[4] = 32;                                            // tile_dim1 = 32 (tile_dim2 = 0)
    g1[5] = HW;                                            // tensor_dim0_stride[31:0]
    g1[6] = 0;
    g1[7] = 0;
    i32x4 gz = {0, 0, 0, 0};
#if __clang_major__ >= 23
    i32x8 gz8 = {0, 0, 0, 0, 0, 0, 0, 0};
    __builtin_amdgcn_tensor_load_to_lds(g0, g1, gz, gz, gz8, 0);
#else
    __builtin_amdgcn_tensor_load_to_lds(g0, g1, gz, gz, 0);
#endif
}
#endif // HAVE_TDM

// ---------------- zero the stats block ----------------
__global__ void zero_stats_kernel(float* stats) {
    int i = blockIdx.x * blockDim.x + threadIdx.x;
    if (i < STATS_FLOATS) stats[i] = 0.0f;
}

// ---------------- 1x1 conv (W[64,256] @ X[256,4096]) + per-(n,o) sum/sumsq ----------------
__global__ void conv_stats_kernel(const float* __restrict__ x, const float* __restrict__ W,
                                  const float* __restrict__ bias, float* __restrict__ Y,
                                  float* __restrict__ sum, float* __restrict__ sumsq) {
    const int blk = blockIdx.x;
    const int n  = blk >> 6;
    const int pb = blk & 63;
    const int o  = threadIdx.x & 63;
    const int pg = threadIdx.x >> 6;
    const int p0 = pb * 64 + pg * 16;

    float acc[16];
    const float bv = bias[o];
#pragma unroll
    for (int i = 0; i < 16; ++i) acc[i] = bv;

    const float* xb = x + (size_t)n * CIN * HW + p0;
    const float* wr = W + o * CIN;
    for (int c = 0; c < CIN; ++c) {
        const float w = wr[c];
        const float* xr = xb + (size_t)c * HW;
#pragma unroll
        for (int i = 0; i < 16; ++i) acc[i] = fmaf(w, xr[i], acc[i]);
    }

    float s = 0.f, ss = 0.f;
    float* yr = Y + ((size_t)n * CQ + o) * HW + p0;
#pragma unroll
    for (int i = 0; i < 16; ++i) { yr[i] = acc[i]; s += acc[i]; ss += acc[i] * acc[i]; }
    atomicAdd(&sum[n * CQ + o], s);
    atomicAdd(&sumsq[n * CQ + o], ss);
}

// ---------------- instance-norm + leaky-relu, accumulate post-activation spatial sum ----------------
__global__ void in_apply_kernel(float* __restrict__ Y, const float* __restrict__ sum,
                                const float* __restrict__ sumsq, float* __restrict__ sumz) {
    const int tid = blockIdx.x * blockDim.x + threadIdx.x;
    const int row = tid >> 8;        // n*64 + o
    const int seg = tid & 255;
    const float m   = sum[row] * (1.0f / HW);
    const float var = sumsq[row] * (1.0f / HW) - m * m;
    const float inv = rsqrtf(var + 1e-5f);
    float* p = Y + (size_t)row * HW + seg * 16;
    float s = 0.f;
#pragma unroll
    for (int i = 0; i < 16; ++i) {
        float z = (p[i] - m) * inv;
        z = (z >= 0.f) ? z : 0.2f * z;
        p[i] = z;
        s += z;
    }
    atomicAdd(&sumz[row], s);
}

// ---------------- spatial-center, per-position channel L2-normalize, pack bf16 position-major ----------------
__global__ void center_norm_pack_kernel(const float* __restrict__ Y, const float* __restrict__ sumz,
                                        __bf16* __restrict__ out) {
    const int tid = blockIdx.x * blockDim.x + threadIdx.x;   // n*4096 total
    const int n   = tid >> 12;
    const int pos = tid & 4095;
    const float* yb = Y + (size_t)n * CQ * HW + pos;
    const float* mz = sumz + n * CQ;
    float ss = 0.f;
#pragma unroll 8
    for (int o = 0; o < CQ; ++o) {
        const float v = yb[(size_t)o * HW] - mz[o] * (1.0f / HW);
        ss += v * v;
    }
    const float inv = 1.0f / (sqrtf(ss) + 1e-5f);
    __bf16* op = out + (size_t)tid * CQ;
#pragma unroll 8
    for (int o = 0; o < CQ; ++o) {
        const float v = yb[(size_t)o * HW] - mz[o] * (1.0f / HW);
        op[o] = f2bf(v * inv);
    }
}

// ---------------- fp32 -> bf16 convert (fc) ----------------
__global__ void cvt_bf16_kernel(const float* __restrict__ in, __bf16* __restrict__ out, int nElem) {
    const int i = blockIdx.x * blockDim.x + threadIdx.x;
    if (i < nElem) out[i] = f2bf(in[i]);
}

// ---------------- energy = ALPHA * fb^T @ fa via v_wmma_f32_16x16x32_bf16 ----------------
// wave computes a 16(k) x 64(q) strip: A(fb) reused over 4 q-tiles; K=64 -> 2 WMMAs/tile.
// Fused per-column sum of exp(E - ALPHA) (fixed max valid: |E| <= ALPHA since columns unit-norm).
__global__ void energy_wmma_kernel(const __bf16* __restrict__ fbp, const __bf16* __restrict__ fap,
                                   float* __restrict__ corr, float* __restrict__ colsum) {
    const int lane   = threadIdx.x & 31;
    const int waveId = threadIdx.x >> 5;
    const int w      = blockIdx.x * 4 + waveId;            // global wave id
    const int n      = w / (256 * 64);
    const int rem    = w % (256 * 64);
    const int kT     = rem / 64;                           // 16-row k tile
    const int qBase  = (rem % 64) * 64;                    // 64-wide q strip
    const int half   = lane >> 4;
    const int l15    = lane & 15;

    // A fragment: fb row (position-major), 16x32 bf16 per chunk
    const __bf16* fbrow = fbp + ((size_t)(n * HW) + kT * 16 + l15) * CQ;
    v16bf A0, A1;
    {
        const int kb0 = half * 8;
        v8bf lo = *(const v8bf*)(fbrow + kb0);
        v8bf hi = *(const v8bf*)(fbrow + kb0 + 16);
#pragma unroll
        for (int i = 0; i < 8; ++i) { A0[i] = lo[i]; A0[i + 8] = hi[i]; }
        const int kb1 = 32 + half * 8;
        lo = *(const v8bf*)(fbrow + kb1);
        hi = *(const v8bf*)(fbrow + kb1 + 16);
#pragma unroll
        for (int i = 0; i < 8; ++i) { A1[i] = lo[i]; A1[i + 8] = hi[i]; }
    }

#pragma unroll
    for (int qt = 0; qt < 4; ++qt) {
        const int q0 = qBase + qt * 16;
        const __bf16* farow = fap + ((size_t)(n * HW) + q0 + l15) * CQ;
        v16bf B0, B1;
        {
            const int b0 = half * 8;
            v8bf lo = *(const v8bf*)(farow + b0);
            v8bf hi = *(const v8bf*)(farow + b0 + 16);
#pragma unroll
            for (int i = 0; i < 8; ++i) { B0[i] = lo[i]; B0[i + 8] = hi[i]; }
            const int b1 = 32 + half * 8;
            lo = *(const v8bf*)(farow + b1);
            hi = *(const v8bf*)(farow + b1 + 16);
#pragma unroll
            for (int i = 0; i < 8; ++i) { B1[i] = lo[i]; B1[i + 8] = hi[i]; }
        }

        v8f c;
#pragma unroll
        for (int i = 0; i < 8; ++i) c[i] = 0.f;
        c = __builtin_amdgcn_wmma_f32_16x16x32_bf16(false, A0, false, B0, (short)0, c, false, false);
        c = __builtin_amdgcn_wmma_f32_16x16x32_bf16(false, A1, false, B1, (short)0, c, false, false);

        // C layout: VGPR r -> row M = r + half*8, col N = lane&15
        const int q = q0 + l15;
        float s = 0.f;
        float* outBase = corr + ((size_t)n * HW + (size_t)(kT * 16 + half * 8)) * HW + q;
#pragma unroll
        for (int r = 0; r < 8; ++r) {
            const float e = c[r] * ALPHA;
            outBase[(size_t)r * HW] = e;
            s += __expf(e - ALPHA);
        }
        s += __shfl_down(s, 16, 32);     // combine row-halves of same column
        if (lane < 16) atomicAdd(&colsum[n * HW + q], s);
    }
}

// ---------------- softmax normalize in place: corr = exp(E - ALPHA) / colsum ----------------
__global__ void softmax_scale_kernel(float* __restrict__ corr, const float* __restrict__ colsum) {
    const size_t stride = (size_t)gridDim.x * blockDim.x;
    const size_t total  = (size_t)NB * HW * HW;
    for (size_t idx = (size_t)blockIdx.x * blockDim.x + threadIdx.x; idx < total; idx += stride) {
        __builtin_prefetch(corr + idx + 4 * stride, 0, 1);   // global_prefetch: streaming pass
        const int n = (int)(idx >> 24);
        const int q = (int)(idx & 4095);
        const float d = colsum[n * HW + q] + 1e-30f;
        corr[idx] = __expf(corr[idx] - ALPHA) / d;
    }
}

// ---------------- warp = fc @ corr via WMMA ----------------
// block = 4 waves; each wave owns one 16-c tile, block shares a 64-q strip; K loop over 4096.
// corr tiles (32k x 64q fp32, 8 KiB) streamed global->LDS by the Tensor Data Mover,
// double-buffered and pipelined on TENSORcnt; bf16 conversion happens at fragment build.
__global__ void warp_wmma_kernel(const __bf16* __restrict__ fcb, const float* __restrict__ corr,
                                 float* __restrict__ warpOut) {
    __shared__ __align__(16) float tiles[2][32][64];   // 16 KiB, [buf][k_local][q_local]

    const int lane   = threadIdx.x & 31;
    const int waveId = threadIdx.x >> 5;
    const int b      = blockIdx.x;                  // NB * 4 * 64 blocks
    const int n      = b / 256;
    const int rem    = b % 256;
    const int cBase  = (rem / 64) * 64 + waveId * 16;
    const int qBase  = (rem % 64) * 64;
    const int half   = lane >> 4;
    const int l15    = lane & 15;
    const int nChunks = HW / 32;                    // 128

    v8f acc[4];
#pragma unroll
    for (int qt = 0; qt < 4; ++qt)
#pragma unroll
        for (int i = 0; i < 8; ++i) acc[qt][i] = 0.f;

    const __bf16* frow  = fcb + ((size_t)(n * CIN) + cBase + l15) * HW;
    const float*  gbase = corr + (size_t)n * HW * HW + qBase;

#ifdef HAVE_TDM
    if (waveId == 0)
        tdm_load_corr_tile(gbase, (unsigned)(size_t)&tiles[0][0][0]);   // prologue: chunk 0
#endif

    for (int i = 0; i < nChunks; ++i) {
        const int k0 = i * 32;
#ifdef HAVE_TDM
        if (waveId == 0) {
            if (i + 1 < nChunks) {
                tdm_load_corr_tile(gbase + (size_t)(k0 + 32) * HW,
                                   (unsigned)(size_t)&tiles[(i + 1) & 1][0][0]);
                __builtin_amdgcn_s_wait_tensorcnt(1);   // chunk i done (TDM in-order per wave)
            } else {
                __builtin_amdgcn_s_wait_tensorcnt(0);
            }
        }
#else
        {   // fallback: cooperative VMEM staging
            const int t  = threadIdx.x;             // 0..127
            const int r  = t >> 2;                  // k row 0..31
            const int qq = (t & 3) << 4;            // 0,16,32,48
            const float* src = gbase + (size_t)(k0 + r) * HW + qq;
#pragma unroll
            for (int j = 0; j < 16; ++j) tiles[i & 1][r][qq + j] = src[j];
        }
#endif
        __syncthreads();                            // tile[i&1] visible to all waves

        // A fragment from fc (row-major [c][k]), 16x32 bf16
        v16bf A;
        {
            const int kb = k0 + half * 16;
            v8bf lo = *(const v8bf*)(frow + kb);
            v8bf hi = *(const v8bf*)(frow + kb + 8);
#pragma unroll
            for (int j = 0; j < 8; ++j) { A[j] = lo[j]; A[j + 8] = hi[j]; }
        }

        const float (*tb)[64] = tiles[i & 1];
#pragma unroll
        for (int qt = 0; qt < 4; ++qt) {
            // B fragment: lane -> column q, 16 k-contiguous values, fp32->bf16 in-register
            const int ql = qt * 16 + l15;
            const float* tp = &tb[half * 16][ql];
            v16bf B;
#pragma unroll
            for (int j = 0; j < 16; ++j) B[j] = f2bf(tp[(size_t)j * 64]);
            acc[qt] = __builtin_amdgcn_wmma_f32_16x16x32_bf16(false, A, false, B, (short)0,
                                                              acc[qt], false, false);
        }
        __syncthreads();                            // all reads done before buffer reuse
    }

#pragma unroll
    for (int qt = 0; qt < 4; ++qt) {
        const int q = qBase + qt * 16 + l15;
        float* outBase = warpOut + ((size_t)n * CIN + cBase + half * 8) * HW + q;
#pragma unroll
        for (int r = 0; r < 8; ++r) outBase[(size_t)r * HW] = acc[qt][r];
    }
}

extern "C" void kernel_launch(void* const* d_in, const int* in_sizes, int n_in,
                              void* d_out, int out_size, void* d_ws, size_t ws_size,
                              hipStream_t stream) {
    (void)in_sizes; (void)n_in; (void)out_size; (void)ws_size;
    const float* fa_raw = (const float*)d_in[0];
    const float* fb_raw = (const float*)d_in[1];
    const float* fc_raw = (const float*)d_in[2];
    const float* Wa     = (const float*)d_in[3];
    const float* ba     = (const float*)d_in[4];
    const float* Wb     = (const float*)d_in[5];
    const float* bb     = (const float*)d_in[6];

    float* warpOut = (float*)d_out;                               // [4,256,64,64]
    float* corr    = (float*)d_out + (size_t)NB * CIN * HW;       // [4,4096,4096]

    char*  ws    = (char*)d_ws;
    float* Ya    = (float*)(ws + OFF_YA);
    float* Yb    = (float*)(ws + OFF_YB);
    __bf16* fap  = (__bf16*)(ws + OFF_FAP);
    __bf16* fbp  = (__bf16*)(ws + OFF_FBP);
    __bf16* fcb  = (__bf16*)(ws + OFF_FCB);
    float* stats = (float*)(ws + OFF_STATS);
    float* sumA   = stats + 0 * NB * CQ;
    float* sumsqA = stats + 1 * NB * CQ;
    float* sumzA  = stats + 2 * NB * CQ;
    float* sumB   = stats + 3 * NB * CQ;
    float* sumsqB = stats + 4 * NB * CQ;
    float* sumzB  = stats + 5 * NB * CQ;
    float* colsum = stats + 6 * NB * CQ;

    zero_stats_kernel<<<(STATS_FLOATS + 255) / 256, 256, 0, stream>>>(stats);

    conv_stats_kernel<<<NB * 64, 256, 0, stream>>>(fa_raw, Wa, ba, Ya, sumA, sumsqA);
    conv_stats_kernel<<<NB * 64, 256, 0, stream>>>(fb_raw, Wb, bb, Yb, sumB, sumsqB);

    in_apply_kernel<<<NB * CQ, 256, 0, stream>>>(Ya, sumA, sumsqA, sumzA);
    in_apply_kernel<<<NB * CQ, 256, 0, stream>>>(Yb, sumB, sumsqB, sumzB);

    center_norm_pack_kernel<<<NB * HW / 256, 256, 0, stream>>>(Ya, sumzA, fap);
    center_norm_pack_kernel<<<NB * HW / 256, 256, 0, stream>>>(Yb, sumzB, fbp);

    cvt_bf16_kernel<<<(NB * CIN * HW) / 256, 256, 0, stream>>>(fc_raw, fcb, NB * CIN * HW);

    // energy: NB * 256 kTiles * 64 qStrips waves, 4 waves/block
    energy_wmma_kernel<<<(NB * 256 * 64) / 4, 128, 0, stream>>>(fbp, fap, corr, colsum);

    softmax_scale_kernel<<<65536, 256, 0, stream>>>(corr, colsum);

    // warp: NB * 4 cGroups * 64 qStrips blocks of 4 waves
    warp_wmma_kernel<<<NB * 4 * 64, 128, 0, stream>>>(fcb, corr, warpOut);
}
